// MixerGatedDeltaNet_50843822850201
// MI455X (gfx1250) — compile-verified, matrix operand order checked by reading, not run on hardware
//
#include <hip/hip_runtime.h>
#include <hip/hip_bf16.h>

// ---------------- problem constants ----------------
#define D_MODEL   1024
#define N_HEADS   16
#define D_HEAD    64
#define HEAD_V    128
#define KEY_DIM   1024            // N_HEADS * D_HEAD
#define VALUE_DIM 2048            // KEY_DIM * 2
#define N_IN      4128            // 2*KEY_DIM + VALUE_DIM + 2*N_HEADS
#define N_IN_PAD  4224            // 33 * 128 (padded for 128-wide N tiles)
#define N_GATE    2048            // N_HEADS * HEAD_V
#define BB        4
#define TT        4096
#define MM        (BB * TT)       // 16384 rows

// ---------------- WMMA types ----------------
typedef __attribute__((ext_vector_type(16))) __bf16 v16bf;
typedef __attribute__((ext_vector_type(8)))  __bf16 v8bf;
typedef __attribute__((ext_vector_type(8)))  float  v8f;

__device__ __forceinline__ unsigned short f2bf(float f) {
  unsigned int u = __float_as_uint(f);
  u += 0x7FFFu + ((u >> 16) & 1u);            // round-to-nearest-even
  return (unsigned short)(u >> 16);
}

__device__ __forceinline__ float sigmoidf_(float x) {
  return 1.0f / (1.0f + __expf(-x));
}

// CDNA5 async global->LDS copy: 16B per lane, tracked by ASYNCcnt.
// LDS byte address = low 32 bits of the generic shared pointer (aperture rule).
__device__ __forceinline__ void async_copy_b128(const unsigned short* g,
                                                unsigned short* l) {
  unsigned loff = (unsigned)(uintptr_t)l;
  unsigned long long ga = (unsigned long long)(uintptr_t)g;
  asm volatile("global_load_async_to_lds_b128 %0, %1, off"
               :: "v"(loff), "v"(ga) : "memory");
}
#define S_WAIT_ASYNCCNT_LE(n) asm volatile("s_wait_asynccnt " #n ::: "memory")

// =====================================================================
// Prep kernel A: hidden_states f32 -> bf16 (row-major [M][K])
// =====================================================================
__global__ __launch_bounds__(256)
void gdn_cvt_bf16(const float* __restrict__ src, unsigned short* __restrict__ dst,
                  long long n) {
  long long i = (long long)blockIdx.x * blockDim.x + threadIdx.x;
  if (i < n) dst[i] = f2bf(src[i]);
}

// =====================================================================
// Prep kernel B: W[K][N] f32 -> WT[Npad][K] bf16 (transposed, zero-padded)
// =====================================================================
__global__ __launch_bounds__(256)
void gdn_prep_wt(const float* __restrict__ W, unsigned short* __restrict__ WT,
                 int N, int Npad, int K) {
  long long i = (long long)blockIdx.x * blockDim.x + threadIdx.x;
  long long total = (long long)Npad * K;
  if (i >= total) return;
  int n = (int)(i / K);
  int k = (int)(i - (long long)n * K);
  float x = (n < N) ? W[(size_t)k * N + n] : 0.0f;
  WT[i] = f2bf(x);
}

// =====================================================================
// Kernel 1: bf16 WMMA GEMM with async global->LDS double buffering.
//   C[M,N] = A[M,K] * B[N,K]^T   (A,B pre-converted bf16, C f32)
// block = 256 threads (8 waves), block tile 128(M) x 128(N), K-step 32.
// wave tile = 32(M) x 64(N) -> 8x v_wmma_f32_16x16x32_bf16 per K-step.
// =====================================================================
#define GT_M 128
#define GT_N 128
#define GT_K 32

__global__ __launch_bounds__(256)
void gdn_gemm_bf16_wmma(const unsigned short* __restrict__ Abf,   // [M][K]
                        const unsigned short* __restrict__ BTbf,  // [Npad][K]
                        float* __restrict__ C,
                        int N, int K) {
  __shared__ __align__(16) unsigned short As[2][GT_M * GT_K];  // [m][k]
  __shared__ __align__(16) unsigned short Bs[2][GT_N * GT_K];  // [n][k]

  const int tid   = threadIdx.x;
  const int wid   = tid >> 5;
  const int lane  = tid & 31;
  const int waveM = wid >> 1;         // 0..3
  const int waveN = wid & 1;          // 0..1
  const int m0    = blockIdx.y * GT_M;
  const int n0    = blockIdx.x * GT_N;

  const int row16 = lane & 15;
  const int kbase = (lane >> 4) * 8;

  v8f acc[2][4];
#pragma unroll
  for (int i = 0; i < 2; ++i)
#pragma unroll
    for (int j = 0; j < 4; ++j)
#pragma unroll
      for (int e = 0; e < 8; ++e) acc[i][j][e] = 0.0f;

  // per-thread async chunks: 512 chunks of 16B per tile, 2 per thread,
  // chunk c -> row = c>>2, k-quarter = (c&3)*8
  const int c0r = tid >> 2,          c0q = (tid & 3) * 8;
  const int c1r = (tid + 256) >> 2,  c1q = ((tid + 256) & 3) * 8;

  // prologue: issue tile 0 (4 async b128 per thread)
  {
    async_copy_b128(Abf  + (size_t)(m0 + c0r) * K + c0q, &As[0][c0r * GT_K + c0q]);
    async_copy_b128(BTbf + (size_t)(n0 + c0r) * K + c0q, &Bs[0][c0r * GT_K + c0q]);
    async_copy_b128(Abf  + (size_t)(m0 + c1r) * K + c1q, &As[0][c1r * GT_K + c1q]);
    async_copy_b128(BTbf + (size_t)(n0 + c1r) * K + c1q, &Bs[0][c1r * GT_K + c1q]);
  }

  const int KT = K / GT_K;
  for (int kt = 0; kt < KT; ++kt) {
    const int buf = kt & 1;
    if (kt + 1 < KT) {
      const int kk = (kt + 1) * GT_K;
      const int nb = buf ^ 1;
      async_copy_b128(Abf  + (size_t)(m0 + c0r) * K + kk + c0q, &As[nb][c0r * GT_K + c0q]);
      async_copy_b128(BTbf + (size_t)(n0 + c0r) * K + kk + c0q, &Bs[nb][c0r * GT_K + c0q]);
      async_copy_b128(Abf  + (size_t)(m0 + c1r) * K + kk + c1q, &As[nb][c1r * GT_K + c1q]);
      async_copy_b128(BTbf + (size_t)(n0 + c1r) * K + kk + c1q, &Bs[nb][c1r * GT_K + c1q]);
      S_WAIT_ASYNCCNT_LE(4);   // in-order: current tile's 4 loads are done
    } else {
      S_WAIT_ASYNCCNT_LE(0);
    }
    __syncthreads();

    // A fragments: 2 x (16x32 bf16), two 16B LDS reads each
    v16bf afrag[2];
#pragma unroll
    for (int i = 0; i < 2; ++i) {
      int r = waveM * 32 + i * 16 + row16;
      v8bf a0 = *reinterpret_cast<const v8bf*>(&As[buf][r * GT_K + kbase]);
      v8bf a1 = *reinterpret_cast<const v8bf*>(&As[buf][r * GT_K + kbase + 16]);
#pragma unroll
      for (int e = 0; e < 8; ++e) { afrag[i][e] = a0[e]; afrag[i][e + 8] = a1[e]; }
    }

#pragma unroll
    for (int j = 0; j < 4; ++j) {
      int n = waveN * 64 + j * 16 + row16;
      v8bf b0 = *reinterpret_cast<const v8bf*>(&Bs[buf][n * GT_K + kbase]);
      v8bf b1 = *reinterpret_cast<const v8bf*>(&Bs[buf][n * GT_K + kbase + 16]);
      v16bf bfrag;
#pragma unroll
      for (int e = 0; e < 8; ++e) { bfrag[e] = b0[e]; bfrag[e + 8] = b1[e]; }
#pragma unroll
      for (int i = 0; i < 2; ++i) {
        acc[i][j] = __builtin_amdgcn_wmma_f32_16x16x32_bf16(
            false, afrag[i], false, bfrag, (short)0, acc[i][j], false, false);
      }
    }
    __syncthreads();   // protect buf before it is refilled two tiles later
  }

  // D layout: lanes 0-15 -> M = r, N = lane ; lanes 16-31 -> M = 8+r, N = lane-16
#pragma unroll
  for (int i = 0; i < 2; ++i) {
    const int mrow = m0 + waveM * 32 + i * 16 + (lane >> 4) * 8;
#pragma unroll
    for (int j = 0; j < 4; ++j) {
      int n = n0 + waveN * 64 + j * 16 + row16;
      if (n < N) {
#pragma unroll
        for (int r = 0; r < 8; ++r)
          C[(size_t)(mrow + r) * N + n] = acc[i][j][r];
      }
    }
  }
}

// =====================================================================
// Kernel 2: causal depthwise conv (K=4) + SiLU for q/k/v projections.
// =====================================================================
__global__ __launch_bounds__(256)
void gdn_conv_silu(const float* __restrict__ qkvba,
                   const float* __restrict__ qw,
                   const float* __restrict__ kw,
                   const float* __restrict__ vw,
                   float* __restrict__ q,
                   float* __restrict__ k,
                   float* __restrict__ v) {
  const long long idx = (long long)blockIdx.x * blockDim.x + threadIdx.x;
  const long long total = (long long)MM * 4096;
  if (idx >= total) return;
  const int ch = (int)(idx & 4095);
  const int bt = (int)(idx >> 12);
  const int t  = bt & (TT - 1);
  const int b  = bt >> 12;

  const float* w;
  if (ch < KEY_DIM)            w = qw + (size_t)ch * 4;
  else if (ch < 2 * KEY_DIM)   w = kw + (size_t)(ch - KEY_DIM) * 4;
  else                         w = vw + (size_t)(ch - 2 * KEY_DIM) * 4;

  float w0 = w[0], w1 = w[1], w2 = w[2], w3 = w[3];
  float acc = 0.0f;
  const size_t base = (size_t)b * TT;
#pragma unroll
  for (int i = 0; i < 4; ++i) {
    int src = t - 3 + i;
    float wi = (i == 0) ? w0 : (i == 1) ? w1 : (i == 2) ? w2 : w3;
    if (src >= 0) acc += qkvba[(base + src) * N_IN + ch] * wi;
  }
  float y = acc * sigmoidf_(acc);   // SiLU

  if (ch < KEY_DIM) {
    int h = ch >> 6, d = ch & 63;
    q[(((size_t)b * N_HEADS + h) * TT + t) * D_HEAD + d] = y;
  } else if (ch < 2 * KEY_DIM) {
    int c = ch - KEY_DIM, h = c >> 6, d = c & 63;
    k[(((size_t)b * N_HEADS + h) * TT + t) * D_HEAD + d] = y;
  } else {
    int c = ch - 2 * KEY_DIM, h = c >> 7, d = c & 127;
    v[(((size_t)b * N_HEADS + h) * TT + t) * HEAD_V + d] = y;
  }
}

// =====================================================================
// Kernel 3: beta = sigmoid(b_proj), g = -exp(A_log) * softplus(a + dt_bias)
// =====================================================================
__global__ __launch_bounds__(256)
void gdn_beta_g(const float* __restrict__ qkvba,
                const float* __restrict__ dt_bias,
                const float* __restrict__ A_log,
                float* __restrict__ beta,
                float* __restrict__ g) {
  const int idx = blockIdx.x * blockDim.x + threadIdx.x;  // B*T*H
  if (idx >= MM * N_HEADS) return;
  const int h  = idx & (N_HEADS - 1);
  const int bt = idx >> 4;
  const int t  = bt & (TT - 1);
  const int b  = bt >> 12;

  const size_t row = (size_t)bt * N_IN;
  float bp = qkvba[row + 2 * KEY_DIM + VALUE_DIM + h];
  float ap = qkvba[row + 2 * KEY_DIM + VALUE_DIM + N_HEADS + h];

  float x = ap + dt_bias[h];
  float sp = (x > 20.0f) ? x : log1pf(__expf(x));   // softplus
  size_t o = ((size_t)b * N_HEADS + h) * TT + t;
  beta[o] = sigmoidf_(bp);
  g[o]    = -__expf(A_log[h]) * sp;
}

// =====================================================================
// Kernel 4: in-place L2 norm of q and k over D_HEAD (q scaled by 1/8).
// =====================================================================
__global__ __launch_bounds__(256)
void gdn_l2norm(float* __restrict__ q, float* __restrict__ k, int nvec) {
  const int idx = blockIdx.x * blockDim.x + threadIdx.x;
  if (idx >= nvec) return;
  float* qp = q + (size_t)idx * D_HEAD;
  float* kp = k + (size_t)idx * D_HEAD;
  float sq = 0.0f, sk = 0.0f;
#pragma unroll
  for (int i = 0; i < D_HEAD; ++i) { float a = qp[i]; sq = fmaf(a, a, sq); }
#pragma unroll
  for (int i = 0; i < D_HEAD; ++i) { float a = kp[i]; sk = fmaf(a, a, sk); }
  float rq = rsqrtf(sq + 1e-6f) * 0.125f;   // * D_HEAD^-0.5
  float rk = rsqrtf(sk + 1e-6f);
#pragma unroll
  for (int i = 0; i < D_HEAD; ++i) qp[i] *= rq;
#pragma unroll
  for (int i = 0; i < D_HEAD; ++i) kp[i] *= rk;
}

// =====================================================================
// Kernel 5: gated delta-rule scan, fused with output gate.
// One block per (b,h); 128 threads; thread owns state column S[0..63].
// =====================================================================
__global__ __launch_bounds__(128)
void gdn_scan(const float* __restrict__ q,
              const float* __restrict__ k,
              const float* __restrict__ v,
              const float* __restrict__ g,
              const float* __restrict__ beta,
              const float* __restrict__ gate,
              float* __restrict__ out) {
  const int bh  = blockIdx.x;           // 0..63
  const int b   = bh >> 4;
  const int h   = bh & 15;
  const int tid = threadIdx.x;          // 0..127 -> V column

  __shared__ float qs[D_HEAD];
  __shared__ float ks[D_HEAD];

  const float* qp = q + (size_t)bh * TT * D_HEAD;
  const float* kp = k + (size_t)bh * TT * D_HEAD;
  const float* vp = v + (size_t)bh * TT * HEAD_V;
  const float* gp = g + (size_t)bh * TT;
  const float* bp = beta + (size_t)bh * TT;

  float S[D_HEAD];
#pragma unroll
  for (int i = 0; i < D_HEAD; ++i) S[i] = 0.0f;

  for (int t = 0; t < TT; ++t) {
    if (tid < D_HEAD) {
      qs[tid] = qp[(size_t)t * D_HEAD + tid];
      ks[tid] = kp[(size_t)t * D_HEAD + tid];
    }
    // pull future rows toward the WGP while the FMA chain runs
    if (t + 16 < TT) {
      __builtin_prefetch(vp + (size_t)(t + 16) * HEAD_V + tid, 0, 1);
      if (tid < D_HEAD)
        __builtin_prefetch(kp + (size_t)(t + 16) * D_HEAD + tid, 0, 1);
    }
    __syncthreads();

    const float vt = vp[(size_t)t * HEAD_V + tid];
    const float gt = gp[t];
    const float bt = bp[t];
    const float eg = __expf(gt);

    // S *= exp(g);  ksum = k . S[:,col]
    float ksum = 0.0f;
#pragma unroll
    for (int i = 0; i < D_HEAD; ++i) {
      S[i] *= eg;
      ksum = fmaf(ks[i], S[i], ksum);
    }
    const float vn = (vt - ksum) * bt;

    // S += k (x) v_new ;  o = q . S[:,col]
    float o = 0.0f;
#pragma unroll
    for (int i = 0; i < D_HEAD; ++i) {
      S[i] = fmaf(ks[i], vn, S[i]);
      o = fmaf(qs[i], S[i], o);
    }

    // fuse output gate: out = o * silu(gate)
    const size_t gi = (size_t)(b * TT + t) * N_GATE + h * HEAD_V + tid;
    const float gv = gate[gi];
    out[gi] = o * gv * sigmoidf_(gv);

    __syncthreads();   // protect qs/ks before next step
  }
}

// =====================================================================
// host launcher
// =====================================================================
extern "C" void kernel_launch(void* const* d_in, const int* in_sizes, int n_in,
                              void* d_out, int out_size, void* d_ws, size_t ws_size,
                              hipStream_t stream) {
  const float* hs      = (const float*)d_in[0];
  const float* W_in    = (const float*)d_in[1];
  const float* q_w     = (const float*)d_in[2];
  const float* k_w     = (const float*)d_in[3];
  const float* v_w     = (const float*)d_in[4];
  const float* dt_bias = (const float*)d_in[5];
  const float* A_log   = (const float*)d_in[6];
  const float* W_gate  = (const float*)d_in[7];
  float* out = (float*)d_out;

  // workspace carve-up
  char* w = (char*)d_ws;
  float* qkvba = (float*)w; w += (size_t)MM * N_IN * 4;
  float* gate  = (float*)w; w += (size_t)MM * N_GATE * 4;
  float* qb    = (float*)w; w += (size_t)BB * N_HEADS * TT * D_HEAD * 4;
  float* kb    = (float*)w; w += (size_t)BB * N_HEADS * TT * D_HEAD * 4;
  float* vb    = (float*)w; w += (size_t)BB * N_HEADS * TT * HEAD_V * 4;
  float* gbuf  = (float*)w; w += (size_t)BB * N_HEADS * TT * 4;
  float* bbuf  = (float*)w; w += (size_t)BB * N_HEADS * TT * 4;
  unsigned short* Abf  = (unsigned short*)w; w += (size_t)MM * D_MODEL * 2;
  unsigned short* WTin = (unsigned short*)w; w += (size_t)N_IN_PAD * D_MODEL * 2;
  unsigned short* WTg  = (unsigned short*)w; w += (size_t)N_GATE * D_MODEL * 2;

  dim3 blk(256);

  // 0) one-time bf16 conversion + weight transpose
  long long nA = (long long)MM * D_MODEL;
  gdn_cvt_bf16<<<(unsigned)((nA + 255) / 256), blk, 0, stream>>>(hs, Abf, nA);
  long long nW1 = (long long)N_IN_PAD * D_MODEL;
  gdn_prep_wt<<<(unsigned)((nW1 + 255) / 256), blk, 0, stream>>>(W_in, WTin, N_IN, N_IN_PAD, D_MODEL);
  long long nW2 = (long long)N_GATE * D_MODEL;
  gdn_prep_wt<<<(unsigned)((nW2 + 255) / 256), blk, 0, stream>>>(W_gate, WTg, N_GATE, N_GATE, D_MODEL);

  // 1) qkvba = hs @ W_in   (WMMA, async double-buffered)
  dim3 g1((N_IN_PAD) / GT_N, MM / GT_M);
  gdn_gemm_bf16_wmma<<<g1, blk, 0, stream>>>(Abf, WTin, qkvba, N_IN, D_MODEL);

  // 2) gate = hs @ W_gate  (WMMA, async double-buffered)
  dim3 g2(N_GATE / GT_N, MM / GT_M);
  gdn_gemm_bf16_wmma<<<g2, blk, 0, stream>>>(Abf, WTg, gate, N_GATE, D_MODEL);

  // 3) conv + SiLU -> q,k,v
  long long convN = (long long)MM * 4096;
  gdn_conv_silu<<<(unsigned)((convN + 255) / 256), blk, 0, stream>>>(
      qkvba, q_w, k_w, v_w, qb, kb, vb);

  // 4) beta / g
  int bgN = MM * N_HEADS;
  gdn_beta_g<<<(bgN + 255) / 256, blk, 0, stream>>>(qkvba, dt_bias, A_log, bbuf, gbuf);

  // 5) l2norm q,k
  int nvec = BB * N_HEADS * TT;
  gdn_l2norm<<<(nvec + 255) / 256, blk, 0, stream>>>(qb, kb, nvec);

  // 6) sequential gated delta scan + output gating
  gdn_scan<<<BB * N_HEADS, dim3(128), 0, stream>>>(qb, kb, vb, gbuf, bbuf, gate, out);
}
